// Net_12532714570361
// MI455X (gfx1250) — compile-verified, matrix operand order checked by reading, not run on hardware
//
#include <hip/hip_runtime.h>
#include <hip/hip_bf16.h>

// ---------------------------------------------------------------------------
// Problem dims
// ---------------------------------------------------------------------------
#define BATCH   8192
#define D_IN    784
#define D_H     1000
#define D_OUT   10
#define T_STEPS 25

#define K1P     800     // D_IN padded to multiple of 32
#define W1ROWS  1024    // D_H padded to multiple of 32 (N-tile coverage)
#define K2P     1024    // D_H padded to multiple of 32
#define W2ROWS  16      // D_OUT padded to 16
#define LD1     1000    // leading dim of cur1 / mem1 (fp32), multiple of 8
#define LD2     16      // leading dim of mem2

#define KT2     (K2P / 32)   // 32 K-tiles in layer-2 GEMM
#define KSPLIT  4            // waves cooperating per 16-row group
#define KT_PER  (KT2 / KSPLIT)

typedef __attribute__((ext_vector_type(16))) _Float16 v16h;
typedef __attribute__((ext_vector_type(8)))  _Float16 v8h;
typedef __attribute__((ext_vector_type(8)))  float    v8f;
typedef __attribute__((ext_vector_type(4)))  float    v4f;

// Load one 16x32 (A) or 32x16 (B) f16 WMMA fragment for this lane from a
// row-major matrix: lane holds 8 contiguous halfs at p and 8 at p+16.
__device__ __forceinline__ v16h load_frag(const _Float16* __restrict__ p) {
  v8h lo = *(const v8h*)(p);
  v8h hi = *(const v8h*)(p + 16);
  v16h r;
#pragma unroll
  for (int i = 0; i < 8; ++i) { r[i] = lo[i]; r[i + 8] = hi[i]; }
  return r;
}

__device__ __forceinline__ v8f wmma_f16(v16h a, v16h b, v8f c) {
  // (neg_a, A, neg_b, B, c_mod, C, reuse_a, reuse_b)
  return __builtin_amdgcn_wmma_f32_16x16x32_f16(false, a, false, b, (short)0, c,
                                                false, false);
}

// ---------------------------------------------------------------------------
// Utility kernels
// ---------------------------------------------------------------------------
__global__ void zero_f32(float* __restrict__ p, int n) {
  int i = blockIdx.x * blockDim.x + threadIdx.x;
  int stride = gridDim.x * blockDim.x;
  for (; i < n; i += stride) p[i] = 0.0f;
}

// fp32 [srows,scols] -> f16 [drows,dcols], zero padding outside source.
__global__ void cvt_pad_f16(const float* __restrict__ src, _Float16* __restrict__ dst,
                            int srows, int scols, int drows, int dcols) {
  int i = blockIdx.x * blockDim.x + threadIdx.x;
  int stride = gridDim.x * blockDim.x;
  int total = drows * dcols;
  for (; i < total; i += stride) {
    int r = i / dcols;
    int k = i - r * dcols;
    float v = (r < srows && k < scols) ? src[r * scols + k] : 0.0f;
    dst[i] = (_Float16)v;
  }
}

// ---------------------------------------------------------------------------
// GEMM1: cur1 = x @ W1^T + b1    ([8192,784]x[1000,784]^T -> [8192,1000] f32)
// One wave computes a 32x32 output block (2 M-frags x 2 N-frags): 4 f32
// accumulators (32 VGPRs) + 4 f16 fragments (32 VGPRs) -> no spills even with
// compiler load pipelining. Base-pointer + kt*32 addressing strength-reduces
// to immediate-offset global_load_b128 pairs.
// ---------------------------------------------------------------------------
__global__ __launch_bounds__(256) void gemm1_wmma(
    const _Float16* __restrict__ xh,
    const _Float16* __restrict__ w1h,
    const float* __restrict__ b1,
    float* __restrict__ cur1) {
  const int tid  = blockIdx.x * blockDim.x + threadIdx.x;
  const int wave = tid >> 5;            // 8192 waves
  const int lane = threadIdx.x & 31;
  const int mt   = wave >> 5;           // 0..255 (32-row tiles)
  const int nt   = wave & 31;           // 0..31  (32-col tiles)
  const int mbase = mt * 32;
  const int nbase = nt * 32;
  const int lrow  = lane & 15;
  const int hsel  = lane >> 4;

  const _Float16* __restrict__ pa = xh  + (size_t)(mbase + lrow) * K1P + hsel * 8;
  const _Float16* __restrict__ pb = w1h + (size_t)(nbase + lrow) * K1P + hsel * 8;

  v8f acc[2][2] = {};

  for (int kt = 0; kt < K1P / 32; ++kt) {
    const int ko = kt * 32;
    v16h a0 = load_frag(pa + ko);
    v16h a1 = load_frag(pa + (size_t)16 * K1P + ko);
    v16h b0 = load_frag(pb + ko);
    v16h b1f = load_frag(pb + (size_t)16 * K1P + ko);
    acc[0][0] = wmma_f16(a0, b0,  acc[0][0]);
    acc[0][1] = wmma_f16(a0, b1f, acc[0][1]);
    acc[1][0] = wmma_f16(a1, b0,  acc[1][0]);
    acc[1][1] = wmma_f16(a1, b1f, acc[1][1]);
  }

  // C layout: lane l, VGPR r -> (M = r + 8*(l>>4), N = l&15) within 16x16 tile
#pragma unroll
  for (int ni = 0; ni < 2; ++ni) {
    const int col = nbase + 16 * ni + lrow;
    if (col < D_H) {
      const float bias = b1[col];
#pragma unroll
      for (int mi = 0; mi < 2; ++mi) {
        const int rbase = mbase + 16 * mi + 8 * hsel;
#pragma unroll
        for (int r = 0; r < 8; ++r)
          cur1[(size_t)(rbase + r) * LD1 + col] = acc[mi][ni][r] + bias;
      }
    }
  }
}

// ---------------------------------------------------------------------------
// Fused 25-step SNN loop, 4-way K-split for occupancy.
// A "group" of KSPLIT=4 waves owns 16 batch rows. Wave kj handles K-tiles
// [kj*8, kj*8+8): it performs the LIF update of its mem1 slice, packs spikes
// directly into the WMMA A-fragment layout (no spk1 buffer), and accumulates a
// partial cur2. Partials are combined through LDS; the kj==0 wave applies the
// layer-2 LIF and records spk2/mem2. Every mem1/mem2 element is owned by
// exactly one lane -> no atomics. cur1+mem1 (~66 MB) stay L2-resident.
// ---------------------------------------------------------------------------
__global__ __launch_bounds__(256) void snn_loop(
    const float* __restrict__ cur1,
    float* __restrict__ mem1,
    float* __restrict__ mem2,
    const _Float16* __restrict__ w2h,
    const float* __restrict__ b2,
    float* __restrict__ out) {
  __shared__ float lacc[8][32][8];       // [wave-in-block][lane][r], 8 KB

  const int lane  = threadIdx.x & 31;
  const int lwave = threadIdx.x >> 5;    // 0..7
  const int gib   = lwave >> 2;          // group-in-block: 0..1
  const int kj    = lwave & (KSPLIT - 1);
  const int group = blockIdx.x * 2 + gib;  // 0..511, owns rows [16g, 16g+16)
  const int lrow  = lane & 15;
  const int hsel  = lane >> 4;
  const int row   = group * 16 + lrow;   // A-fragment row for this lane
  const int n     = lrow;                // output column (D_OUT padded to 16)
  const float b2n = (n < D_OUT) ? b2[n] : 0.0f;

  float* __restrict__ out_spk = out;
  float* __restrict__ out_mem = out + (size_t)T_STEPS * BATCH * D_OUT;

  const float* __restrict__ crow = cur1 + (size_t)row * LD1;
  float* __restrict__ mrow       = mem1 + (size_t)row * LD1;
  const _Float16* __restrict__ brow = w2h + (size_t)n * K2P + hsel * 8;

  for (int t = 0; t < T_STEPS; ++t) {
    v8f acc = {};
    for (int kt = kj * KT_PER; kt < (kj + 1) * KT_PER; ++kt) {
      const int kbase = kt * 32 + hsel * 8;
      v16h a;
#pragma unroll
      for (int run = 0; run < 2; ++run) {
        const int ks = kbase + run * 16;      // 8-aligned; D_H % 8 == 0
        if (ks < D_H) {
          v4f c0 = *(const v4f*)(crow + ks);
          v4f c1 = *(const v4f*)(crow + ks + 4);
          v4f m0 = *(const v4f*)(mrow + ks);
          v4f m1 = *(const v4f*)(mrow + ks + 4);
#pragma unroll
          for (int j = 0; j < 4; ++j) {
            float mv = m0[j];
            float rst = (mv > 1.0f) ? 1.0f : 0.0f;  // reset from incoming mem
            mv = 0.95f * mv + c0[j] - rst;          // beta*mem + cur - reset*thr
            m0[j] = mv;
            a[run * 8 + j] = (_Float16)((mv > 1.0f) ? 1.0f : 0.0f);
          }
#pragma unroll
          for (int j = 0; j < 4; ++j) {
            float mv = m1[j];
            float rst = (mv > 1.0f) ? 1.0f : 0.0f;
            mv = 0.95f * mv + c1[j] - rst;
            m1[j] = mv;
            a[run * 8 + 4 + j] = (_Float16)((mv > 1.0f) ? 1.0f : 0.0f);
          }
          *(v4f*)(mrow + ks) = m0;
          *(v4f*)(mrow + ks + 4) = m1;
        } else {
#pragma unroll
          for (int j = 0; j < 8; ++j) a[run * 8 + j] = (_Float16)0.0f;
        }
      }
      v16h b = load_frag(brow + kt * 32);      // B[k][n] = W2[n][k]
      acc = wmma_f16(a, b, acc);
    }

    // Combine the 4 partial accumulators of this group via LDS.
    if (kj != 0) {
#pragma unroll
      for (int r = 0; r < 8; ++r) lacc[lwave][lane][r] = acc[r];
    }
    __syncthreads();

    if (kj == 0) {
#pragma unroll
      for (int r = 0; r < 8; ++r)
        acc[r] += lacc[gib * 4 + 1][lane][r] +
                  lacc[gib * 4 + 2][lane][r] +
                  lacc[gib * 4 + 3][lane][r];

      // Layer-2 LIF + record outputs. Lane owns (m, n) for r = 0..7.
#pragma unroll
      for (int r = 0; r < 8; ++r) {
        const int m = group * 16 + hsel * 8 + r;
        float* __restrict__ m2p = mem2 + (size_t)m * LD2 + n;
        float mv = *m2p;
        float rst = (mv > 1.0f) ? 1.0f : 0.0f;
        mv = 0.95f * mv + (acc[r] + b2n) - rst;
        *m2p = mv;
        if (n < D_OUT) {
          const size_t oi = ((size_t)t * BATCH + m) * D_OUT + n;
          out_spk[oi] = (mv > 1.0f) ? 1.0f : 0.0f;
          out_mem[oi] = mv;
        }
      }
    }
    __syncthreads();   // protect lacc before next step overwrites it
  }
}

// ---------------------------------------------------------------------------
// Launch
// ---------------------------------------------------------------------------
extern "C" void kernel_launch(void* const* d_in, const int* in_sizes, int n_in,
                              void* d_out, int out_size, void* d_ws, size_t ws_size,
                              hipStream_t stream) {
  (void)in_sizes; (void)n_in; (void)out_size; (void)ws_size;

  const float* x  = (const float*)d_in[0];   // [8192, 784]
  const float* W1 = (const float*)d_in[1];   // [1000, 784]
  const float* b1 = (const float*)d_in[2];   // [1000]
  const float* W2 = (const float*)d_in[3];   // [10, 1000]
  const float* b2 = (const float*)d_in[4];   // [10]
  float* out = (float*)d_out;                // spk2_rec | mem2_rec, each [25,8192,10]

  // Workspace layout (bytes)
  const size_t CUR1_B = (size_t)BATCH * LD1 * 4;          // 32,768,000
  const size_t MEM1_B = (size_t)BATCH * LD1 * 4;          // 32,768,000
  const size_t MEM2_B = (size_t)BATCH * LD2 * 4;          //    524,288
  const size_t XH_B   = (size_t)BATCH * K1P * 2;          // 13,107,200
  const size_t W1H_B  = (size_t)W1ROWS * K1P * 2;         //  1,638,400

  char* ws = (char*)d_ws;
  float*    cur1 = (float*)(ws);
  float*    mem1 = (float*)(ws + CUR1_B);
  float*    mem2 = (float*)(ws + CUR1_B + MEM1_B);
  _Float16* xh   = (_Float16*)(ws + CUR1_B + MEM1_B + MEM2_B);
  _Float16* w1h  = (_Float16*)(ws + CUR1_B + MEM1_B + MEM2_B + XH_B);
  _Float16* w2h  = (_Float16*)(ws + CUR1_B + MEM1_B + MEM2_B + XH_B + W1H_B);

  // mem1 and mem2 are contiguous -> one zero-init pass
  zero_f32<<<512, 256, 0, stream>>>(mem1, (int)((MEM1_B + MEM2_B) / 4));

  cvt_pad_f16<<<2048, 256, 0, stream>>>(x,  xh,  BATCH, D_IN, BATCH,  K1P);
  cvt_pad_f16<<<256,  256, 0, stream>>>(W1, w1h, D_H,   D_IN, W1ROWS, K1P);
  cvt_pad_f16<<<16,   256, 0, stream>>>(W2, w2h, D_OUT, D_H,  W2ROWS, K2P);

  // 8192 waves (256 M-tiles x 32 N-tiles), 8 waves per 256-thread block
  gemm1_wmma<<<1024, 256, 0, stream>>>(xh, w1h, b1, cur1);

  // 512 groups x 4 K-split waves = 2048 waves (2 groups per 256-thread block)
  snn_loop<<<256, 256, 0, stream>>>(cur1, mem1, mem2, w2h, b2, out);
}
